// LocalAttention_66228395705287
// MI455X (gfx1250) — matrix operand order repaired and checked by reading, not compiled, not run on hardware
//
#include <hip/hip_runtime.h>
#include <hip/hip_bf16.h>

static constexpr int SEQ    = 65536;
static constexpr int DMODEL = 512;
static constexpr int NH     = 8;
static constexpr int WSZ    = 64;

typedef __attribute__((ext_vector_type(16))) __bf16 bf16x16;
typedef __attribute__((ext_vector_type(8)))  float  f32x8;

union FragU { bf16x16 v; uint4 q[2]; };

#if defined(__HIP_DEVICE_COMPILE__) && __has_builtin(__builtin_amdgcn_global_load_async_to_lds_b128)
#define ASYNC_LDS_COPY 1
// exact param types from hipcc's diagnostic: vector_size(16) int, AS1 src / AS3 dst
typedef int v4i_t __attribute__((vector_size(16)));
typedef __attribute__((address_space(1))) v4i_t v4i_global;
typedef __attribute__((address_space(3))) v4i_t v4i_lds;
#endif

// A-matrix 16x32 bf16 fragment (ISA 7.12.2): lane half selects K 0-7/8-15 then 16-23/24-31
__device__ __forceinline__ bf16x16 load_frag_a(const __bf16* lds, int row_base, int ldm, int kbase) {
    int lane = threadIdx.x & 31;
    int r = lane & 15;
    int h = lane >> 4;
    const __bf16* p = lds + (row_base + r) * ldm + kbase + h * 8;
    FragU f;
    f.q[0] = *(const uint4*)p;
    f.q[1] = *(const uint4*)(p + 16);
    return f.v;
}

// B-matrix 32x16 bf16 fragment: lane = column, lane half selects K 0-7/8-15 then 16-23/24-31
__device__ __forceinline__ bf16x16 load_frag_b(const __bf16* lds, int row_base, int ldm, int kbase) {
    int lane = threadIdx.x & 31;
    int n = lane & 15;
    int h = lane >> 4;
    const __bf16* p = lds + (row_base + n) * ldm + kbase + h * 8;
    FragU f;
    f.q[0] = *(const uint4*)p;
    f.q[1] = *(const uint4*)(p + 16);
    return f.v;
}

#define WMMA_BF16(a, b, c) \
    __builtin_amdgcn_wmma_f32_16x16x32_bf16(false, (a), false, (b), (short)0, (c), false, false)

// ---------------- Kernel 1: 6 fused input projections, fp32 in -> bf16 out -------------
// grid (4 n-tiles, 512 m-tiles, 6 projections), block 256 (8 waves)
// double-buffered LDS: one barrier per k-step, next-step global loads overlap WMMAs
__global__ __launch_bounds__(256) void proj_kernel(
    const float* __restrict__ h, const float* __restrict__ h1,
    const float* __restrict__ W, const float* __restrict__ Bias,
    __bf16* __restrict__ qkv)
{
    __shared__ __bf16 Xs[2][128 * 40];
    __shared__ __bf16 Wls[2][128 * 40];

    int pj = blockIdx.z;
    const float* X  = (pj < 3) ? h : h1;
    const float* Wm = W + (size_t)pj * DMODEL * DMODEL;
    const float* Bv = Bias + (size_t)pj * DMODEL;
    __bf16* Out     = qkv + (size_t)pj * SEQ * DMODEL;

    int tid  = threadIdx.x;
    int wave = tid >> 5, lane = tid & 31;
    int wm = wave >> 2, wn = wave & 3;
    int n0 = blockIdx.x * 128, m0 = blockIdx.y * 128;

    int lr = tid >> 1, lcb = (tid & 1) * 16;
    const float* xrow = X  + (size_t)(m0 + lr) * DMODEL + lcb;
    const float* wrow = Wm + (size_t)(n0 + lr) * DMODEL + lcb;

    f32x8 acc[4][2] = {};
    float4 fx[4], fw[4];

    auto load_regs = [&](int k0) {
#pragma unroll
        for (int i = 0; i < 4; i++) {
            fx[i] = *(const float4*)(xrow + k0 + i * 4);
            fw[i] = *(const float4*)(wrow + k0 + i * 4);
        }
        if (k0 + 32 < DMODEL) {                       // prime L2 one more step ahead
            __builtin_prefetch(xrow + k0 + 32, 0, 0);
            __builtin_prefetch(wrow + k0 + 32, 0, 0);
        }
    };
    auto store_lds = [&](int buf) {
#pragma unroll
        for (int i = 0; i < 4; i++) {
            __bf16* dx = &Xs[buf][lr * 40 + lcb + i * 4];
            dx[0] = (__bf16)fx[i].x; dx[1] = (__bf16)fx[i].y;
            dx[2] = (__bf16)fx[i].z; dx[3] = (__bf16)fx[i].w;
            __bf16* dw = &Wls[buf][lr * 40 + lcb + i * 4];
            dw[0] = (__bf16)fw[i].x; dw[1] = (__bf16)fw[i].y;
            dw[2] = (__bf16)fw[i].z; dw[3] = (__bf16)fw[i].w;
        }
    };

    load_regs(0);
    store_lds(0);
    __syncthreads();

    int buf = 0;
    for (int k0 = 0; k0 < DMODEL; k0 += 32) {
        bool more = (k0 + 32) < DMODEL;
        if (more) load_regs(k0 + 32);

        bf16x16 af[4], bf[2];
#pragma unroll
        for (int mi = 0; mi < 4; mi++) af[mi] = load_frag_a(Xs[buf], wm * 64 + mi * 16, 40, 0);
#pragma unroll
        for (int ni = 0; ni < 2; ni++) bf[ni] = load_frag_b(Wls[buf], wn * 32 + ni * 16, 40, 0);
#pragma unroll
        for (int mi = 0; mi < 4; mi++)
#pragma unroll
            for (int ni = 0; ni < 2; ni++)
                acc[mi][ni] = WMMA_BF16(af[mi], bf[ni], acc[mi][ni]);

        if (more) {
            store_lds(buf ^ 1);
            __syncthreads();
            buf ^= 1;
        }
    }

#pragma unroll
    for (int ni = 0; ni < 2; ni++) {
        int col = n0 + wn * 32 + ni * 16 + (lane & 15);
        float bb = Bv[col];
#pragma unroll
        for (int mi = 0; mi < 4; mi++) {
            int rb = m0 + wm * 64 + mi * 16 + (lane >> 4) * 8;
#pragma unroll
            for (int e = 0; e < 8; e++)
                Out[(size_t)(rb + e) * DMODEL + col] = (__bf16)(acc[mi][ni][e] + bb);
        }
    }
}

// ---------------- Kernel 2: windowed attention, bf16 in/out, fp32 softmax --------------
// grid (1024 windows, 8 heads, 2 passes), block 128 (4 waves, one 16-row strip each)
__global__ __launch_bounds__(128) void attn_kernel(
    const __bf16* __restrict__ qkv, __bf16* __restrict__ attO)
{
    __shared__ __bf16 Qs[64 * 72];
    __shared__ __bf16 Ks[64 * 72];
    __shared__ __bf16 Vt[64 * 72];
    __shared__ __bf16 Ps[64 * 72];

    int win = blockIdx.x, head = blockIdx.y, pass = blockIdx.z;
    // pass 0: Q attends Ks/Vs ; pass 1: Qs attends K/V
    int qi = pass ? 3 : 0;
    int ki = pass ? 1 : 4;
    int vi = pass ? 2 : 5;

    int tid = threadIdx.x;
    int r = tid >> 1, hf = tid & 1;

    size_t rowoff = (size_t)(win * WSZ + r) * DMODEL + head * 64 + hf * 32;
    const uint4* qg = (const uint4*)(qkv + (size_t)qi * SEQ * DMODEL + rowoff);
    const uint4* kg = (const uint4*)(qkv + (size_t)ki * SEQ * DMODEL + rowoff);
    const uint4* vg = (const uint4*)(qkv + (size_t)vi * SEQ * DMODEL + rowoff);
    uint4* qd = (uint4*)&Qs[r * 72 + hf * 32];
    uint4* kd = (uint4*)&Ks[r * 72 + hf * 32];

#if ASYNC_LDS_COPY
    // bf16 tiles need no conversion: use the CDNA5 async global->LDS path (ASYNCcnt)
#pragma unroll
    for (int i = 0; i < 4; i++) {
        __builtin_amdgcn_global_load_async_to_lds_b128(
            (v4i_global*)(qg + i), (v4i_lds*)(qd + i), 0, 0);
        __builtin_amdgcn_global_load_async_to_lds_b128(
            (v4i_global*)(kg + i), (v4i_lds*)(kd + i), 0, 0);
    }
#else
#pragma unroll
    for (int i = 0; i < 4; i++) { qd[i] = qg[i]; kd[i] = kg[i]; }
#endif

    uint4 vv[4];
#pragma unroll
    for (int i = 0; i < 4; i++) vv[i] = vg[i];
    const unsigned short* sp = (const unsigned short*)vv;
    unsigned short* VtU = (unsigned short*)Vt;    // V stored transposed: Vt[dk][seq]
#pragma unroll
    for (int e = 0; e < 32; e++) VtU[(hf * 32 + e) * 72 + r] = sp[e];

#if ASYNC_LDS_COPY
#if __has_builtin(__builtin_amdgcn_s_wait_asynccnt)
    __builtin_amdgcn_s_wait_asynccnt(0);
#else
    asm volatile("s_wait_asynccnt 0x0" ::: "memory");
#endif
#endif
    __syncthreads();

    int wave = tid >> 5, lane = tid & 31;
    int mr = wave * 16;

    // S = Q * K^T  (K rows are columns of B; row-major K tile feeds B frags directly)
    f32x8 s[4] = {};
#pragma unroll
    for (int kk = 0; kk < 64; kk += 32) {
        bf16x16 a = load_frag_a(Qs, mr, 72, kk);
#pragma unroll
        for (int nt = 0; nt < 4; nt++) {
            bf16x16 b = load_frag_b(Ks, nt * 16, 72, kk);
            s[nt] = WMMA_BF16(a, b, s[nt]);
        }
    }

    // softmax over 64 cols: 4 per-lane values + 16-lane half-wave reductions
    const float sc = 0.125f;   // 1/sqrt(DK)
#pragma unroll
    for (int p = 0; p < 8; p++) {
        float v0 = s[0][p] * sc, v1 = s[1][p] * sc, v2 = s[2][p] * sc, v3 = s[3][p] * sc;
        float m = fmaxf(fmaxf(v0, v1), fmaxf(v2, v3));
#pragma unroll
        for (int msk = 1; msk < 16; msk <<= 1) m = fmaxf(m, __shfl_xor(m, msk, 32));
        float e0 = __expf(v0 - m), e1 = __expf(v1 - m), e2 = __expf(v2 - m), e3 = __expf(v3 - m);
        float sum = e0 + e1 + e2 + e3;
#pragma unroll
        for (int msk = 1; msk < 16; msk <<= 1) sum += __shfl_xor(sum, msk, 32);
        float inv = 1.0f / sum;
        int row = mr + p + (lane >> 4) * 8;
        int cb  = lane & 15;
        Ps[row * 72 + cb]      = (__bf16)(e0 * inv);
        Ps[row * 72 + 16 + cb] = (__bf16)(e1 * inv);
        Ps[row * 72 + 32 + cb] = (__bf16)(e2 * inv);
        Ps[row * 72 + 48 + cb] = (__bf16)(e3 * inv);
    }
    // each wave only reads back its own 16 P rows: same-wave LDS is in-order, no barrier

    // O = P * V   (Vt row-major per dk column feeds B frags contiguously)
    f32x8 o[4] = {};
#pragma unroll
    for (int kk = 0; kk < 64; kk += 32) {
        bf16x16 a = load_frag_a(Ps, mr, 72, kk);
#pragma unroll
        for (int nt = 0; nt < 4; nt++) {
            bf16x16 b = load_frag_b(Vt, nt * 16, 72, kk);
            o[nt] = WMMA_BF16(a, b, o[nt]);
        }
    }

    __bf16* Og = attO + (size_t)pass * SEQ * DMODEL;
#pragma unroll
    for (int nt = 0; nt < 4; nt++)
#pragma unroll
        for (int p = 0; p < 8; p++) {
            int row = mr + p + (lane >> 4) * 8;
            int col = nt * 16 + (lane & 15);
            Og[(size_t)(win * WSZ + row) * DMODEL + head * 64 + col] = (__bf16)(o[nt][p]);
        }
}

// ---------------- Kernel 3: 2 output projections, bf16 in -> fp32 out ------------------
// grid (4 n-tiles, 512 m-tiles, 2 passes), block 256, double-buffered LDS
__global__ __launch_bounds__(256) void outproj_kernel(
    const __bf16* __restrict__ attO, const float* __restrict__ W,
    const float* __restrict__ Bias, float* __restrict__ OutAll)
{
    __shared__ __bf16 Xs[2][128 * 40];
    __shared__ __bf16 Wls[2][128 * 40];

    int pz = blockIdx.z;
    const __bf16* X = attO + (size_t)pz * SEQ * DMODEL;
    const float* Wm = W + (size_t)(6 + pz) * DMODEL * DMODEL;
    const float* Bv = Bias + (size_t)(6 + pz) * DMODEL;
    float* Out      = OutAll + (size_t)pz * SEQ * DMODEL;

    int tid  = threadIdx.x;
    int wave = tid >> 5, lane = tid & 31;
    int wm = wave >> 2, wn = wave & 3;
    int n0 = blockIdx.x * 128, m0 = blockIdx.y * 128;

    int lr = tid >> 1, lcb = (tid & 1) * 16;
    const __bf16* xrow = X  + (size_t)(m0 + lr) * DMODEL + lcb;
    const float*  wrow = Wm + (size_t)(n0 + lr) * DMODEL + lcb;

    f32x8 acc[4][2] = {};
    uint4  xr[2];
    float4 fw[4];

    auto load_regs = [&](int k0) {
        const uint4* sx = (const uint4*)(xrow + k0);
        xr[0] = sx[0];
        xr[1] = sx[1];
#pragma unroll
        for (int i = 0; i < 4; i++) fw[i] = *(const float4*)(wrow + k0 + i * 4);
        if (k0 + 32 < DMODEL) {
            __builtin_prefetch(xrow + k0 + 32, 0, 0);
            __builtin_prefetch(wrow + k0 + 32, 0, 0);
        }
    };
    auto store_lds = [&](int buf) {
        *(uint4*)&Xs[buf][lr * 40 + lcb]     = xr[0];
        *(uint4*)&Xs[buf][lr * 40 + lcb + 8] = xr[1];
#pragma unroll
        for (int i = 0; i < 4; i++) {
            __bf16* dw = &Wls[buf][lr * 40 + lcb + i * 4];
            dw[0] = (__bf16)fw[i].x; dw[1] = (__bf16)fw[i].y;
            dw[2] = (__bf16)fw[i].z; dw[3] = (__bf16)fw[i].w;
        }
    };

    load_regs(0);
    store_lds(0);
    __syncthreads();

    int buf = 0;
    for (int k0 = 0; k0 < DMODEL; k0 += 32) {
        bool more = (k0 + 32) < DMODEL;
        if (more) load_regs(k0 + 32);

        bf16x16 af[4], bf[2];
#pragma unroll
        for (int mi = 0; mi < 4; mi++) af[mi] = load_frag_a(Xs[buf], wm * 64 + mi * 16, 40, 0);
#pragma unroll
        for (int ni = 0; ni < 2; ni++) bf[ni] = load_frag_b(Wls[buf], wn * 32 + ni * 16, 40, 0);
#pragma unroll
        for (int mi = 0; mi < 4; mi++)
#pragma unroll
            for (int ni = 0; ni < 2; ni++)
                acc[mi][ni] = WMMA_BF16(af[mi], bf[ni], acc[mi][ni]);

        if (more) {
            store_lds(buf ^ 1);
            __syncthreads();
            buf ^= 1;
        }
    }

#pragma unroll
    for (int ni = 0; ni < 2; ni++) {
        int col = n0 + wn * 32 + ni * 16 + (lane & 15);
        float bb = Bv[col];
#pragma unroll
        for (int mi = 0; mi < 4; mi++) {
            int rb = m0 + wm * 64 + mi * 16 + (lane >> 4) * 8;
#pragma unroll
            for (int e = 0; e < 8; e++)
                Out[(size_t)(rb + e) * DMODEL + col] = acc[mi][ni][e] + bb;
        }
    }
}

extern "C" void kernel_launch(void* const* d_in, const int* in_sizes, int n_in,
                              void* d_out, int out_size, void* d_ws, size_t ws_size,
                              hipStream_t stream) {
    const float* h  = (const float*)d_in[0];
    const float* h1 = (const float*)d_in[1];
    const float* W  = (const float*)d_in[2];
    const float* Bv = (const float*)d_in[3];

    // workspace: 6 projections (Q,K,V,Qs,Ks,Vs) + 2 attention outputs, all bf16
    __bf16* qkv  = (__bf16*)d_ws;
    __bf16* attO = qkv + (size_t)6 * SEQ * DMODEL;

    proj_kernel<<<dim3(4, 512, 6), 256, 0, stream>>>(h, h1, W, Bv, qkv);
    attn_kernel<<<dim3(SEQ / WSZ, NH, 2), 128, 0, stream>>>(qkv, attO);
    outproj_kernel<<<dim3(4, 512, 2), 256, 0, stream>>>(attO, W, Bv, (float*)d_out);
}
